// MHAAttention_7499012899639
// MI455X (gfx1250) — matrix operand order repaired and checked by reading, not compiled, not run on hardware
//
#include <hip/hip_runtime.h>
#include <hip/hip_bf16.h>
#include <math.h>

typedef __attribute__((ext_vector_type(16))) _Float16 v16h;
typedef __attribute__((ext_vector_type(8)))  _Float16 v8h;
typedef __attribute__((ext_vector_type(4)))  _Float16 v4h;
typedef __attribute__((ext_vector_type(8)))  float    v8f;
typedef __attribute__((ext_vector_type(4)))  int      i4v;

#define B_  2
#define N_  4096
#define H_  8
#define D_  64
#define HD_ 512
#define LPAD 72   // padded LDS row stride in halves (144B = 36 banks, conflict-free b128)

// ---- gfx1250 async-copy path (ASYNCcnt), guarded so either toolchain compiles ----
#if __has_builtin(__builtin_amdgcn_global_load_async_to_lds_b128) && \
    __has_builtin(__builtin_amdgcn_s_wait_asynccnt)
#define USE_ASYNC 1
#endif

__device__ __forceinline__ void stage16(_Float16* l, const _Float16* g) {
#ifdef USE_ASYNC
    __builtin_amdgcn_global_load_async_to_lds_b128(
        (__attribute__((address_space(1))) i4v*)g,
        (__attribute__((address_space(3))) i4v*)l, 0, 0);
#else
    *(v8h*)l = *(const v8h*)g;
#endif
}

__device__ __forceinline__ void stage_wait() {
#ifdef USE_ASYNC
    __builtin_amdgcn_s_wait_asynccnt(0);
#endif
}

// ---- 16-lane butterfly reductions: DPP (pure VALU) with shfl fallback ----
#if __has_builtin(__builtin_amdgcn_update_dpp)
#define USE_DPP 1
template <int CTRL>
__device__ __forceinline__ float dppmov(float x) {
    int xi = __builtin_bit_cast(int, x);
    int r  = __builtin_amdgcn_update_dpp(xi, xi, CTRL, 0xf, 0xf, true);
    return __builtin_bit_cast(float, r);
}
#endif

// quad_perm[1,0,3,2]=0xB1 (xor1), quad_perm[2,3,0,1]=0x4E (xor2),
// row_ror:4=0x124, row_ror:8=0x128 -- ror4+ror8 folds all 4 quads into every lane.
__device__ __forceinline__ float red16_max(float x) {
#ifdef USE_DPP
    x = fmaxf(x, dppmov<0xB1>(x));
    x = fmaxf(x, dppmov<0x4E>(x));
    x = fmaxf(x, dppmov<0x124>(x));
    x = fmaxf(x, dppmov<0x128>(x));
#else
    x = fmaxf(x, __shfl_xor(x, 1, 32));
    x = fmaxf(x, __shfl_xor(x, 2, 32));
    x = fmaxf(x, __shfl_xor(x, 4, 32));
    x = fmaxf(x, __shfl_xor(x, 8, 32));
#endif
    return x;
}

__device__ __forceinline__ float red16_sum(float x) {
#ifdef USE_DPP
    x += dppmov<0xB1>(x);
    x += dppmov<0x4E>(x);
    x += dppmov<0x124>(x);
    x += dppmov<0x128>(x);
#else
    x += __shfl_xor(x, 1, 32);
    x += __shfl_xor(x, 2, 32);
    x += __shfl_xor(x, 4, 32);
    x += __shfl_xor(x, 8, 32);
#endif
    return x;
}

__device__ __forceinline__ v8f wmma16(v16h a, v16h b, v8f c) {
    return __builtin_amdgcn_wmma_f32_16x16x32_f16(
        false, a, false, b, (short)0, c, false, false);
}

// A-fragment (16x32 f16) from a global f32 row. Elements 0..15 hold K = kb+group*16..+15
// (contiguous, matching B fragments).
__device__ __forceinline__ v16h lda_g(const float* rowp, int kb, int group, float scale) {
    const float* p0 = rowp + kb + group * 16;
    const float* p1 = p0 + 8;
    float4 x0 = *(const float4*)(p0);
    float4 x1 = *(const float4*)(p0 + 4);
    float4 y0 = *(const float4*)(p1);
    float4 y1 = *(const float4*)(p1 + 4);
    v16h a;
    a[0]  = (_Float16)(x0.x * scale); a[1]  = (_Float16)(x0.y * scale);
    a[2]  = (_Float16)(x0.z * scale); a[3]  = (_Float16)(x0.w * scale);
    a[4]  = (_Float16)(x1.x * scale); a[5]  = (_Float16)(x1.y * scale);
    a[6]  = (_Float16)(x1.z * scale); a[7]  = (_Float16)(x1.w * scale);
    a[8]  = (_Float16)(y0.x * scale); a[9]  = (_Float16)(y0.y * scale);
    a[10] = (_Float16)(y0.z * scale); a[11] = (_Float16)(y0.w * scale);
    a[12] = (_Float16)(y1.x * scale); a[13] = (_Float16)(y1.y * scale);
    a[14] = (_Float16)(y1.z * scale); a[15] = (_Float16)(y1.w * scale);
    return a;
}

// A-fragment from an LDS f16 row (same element->K mapping).
__device__ __forceinline__ v16h lda_l(const _Float16* rowp, int kb, int group) {
    v8h s0 = *(const v8h*)(rowp + kb + group * 16);
    v8h s1 = *(const v8h*)(rowp + kb + 8 + group * 16);
    v16h a;
#pragma unroll
    for (int i = 0; i < 8; ++i) { a[i] = s0[i]; a[i + 8] = s1[i]; }
    return a;
}

// B-fragment (32x16 f16): lane holds col n = lane%16, K = group*16 + e sequential.
__device__ __forceinline__ v16h ldb_l(const _Float16* p) {
    v8h lo = *(const v8h*)(p);
    v8h hi = *(const v8h*)(p + 8);
    v16h b;
#pragma unroll
    for (int i = 0; i < 8; ++i) { b[i] = lo[i]; b[i + 8] = hi[i]; }
    return b;
}

// ---------------------------------------------------------------------------
// Kernel 0: one-time precision/layout prep.
//   Kh [b][h][n][d]  f16   (contiguous 64-dim rows per (b,h))
//   Vt [b][h][d][n]  f16   (pre-transposed so PV B-fragments are contiguous)
// ---------------------------------------------------------------------------
__global__ __launch_bounds__(256) void mha_cvt_kernel(
    const float* __restrict__ K, const float* __restrict__ V,
    _Float16* __restrict__ Kh, _Float16* __restrict__ Vt) {
    const size_t e  = ((size_t)blockIdx.x * 256 + threadIdx.x) * 4; // f32 element index
    const int    hd = (int)(e & 511);
    const size_t bn = e >> 9;
    const int    n  = (int)(bn & (N_ - 1));
    const int    b  = (int)(bn >> 12);
    const int    h  = hd >> 6;
    const int    d  = hd & 63;

    float4 kf = *(const float4*)(K + e);
    float4 vf = *(const float4*)(V + e);

    v4h kh;
    kh[0] = (_Float16)kf.x; kh[1] = (_Float16)kf.y;
    kh[2] = (_Float16)kf.z; kh[3] = (_Float16)kf.w;
    *(v4h*)(Kh + (((size_t)(b * H_ + h)) * N_ + n) * D_ + d) = kh;

    _Float16* vbase = Vt + ((size_t)(b * H_ + h) * D_ + d) * N_ + n;
    vbase[0 * N_] = (_Float16)vf.x;
    vbase[1 * N_] = (_Float16)vf.y;
    vbase[2 * N_] = (_Float16)vf.z;
    vbase[3 * N_] = (_Float16)vf.w;
}

// ---------------------------------------------------------------------------
// Kernel 1: flash attention. Block = one (b,h) x 64 query rows; 4 waves x 16 rows.
// Double-buffered LDS: async copies for chunk nc+1 overlap WMMA on chunk nc.
// ---------------------------------------------------------------------------
__global__ __launch_bounds__(256) void mha_flash_attn_kernel(
    const float* __restrict__ Q, const _Float16* __restrict__ Kh,
    const _Float16* __restrict__ Vth, float* __restrict__ ctx) {
    __shared__ _Float16 Kl[2][64 * LPAD];       // Kl[buf][key][dim]
    __shared__ _Float16 Vl[2][64 * LPAD];       // Vl[buf][dim][key]
    __shared__ _Float16 Pl[4 * 16 * LPAD];      // per-wave P staging [row][key]

    const int tid   = threadIdx.x;
    const int wave  = tid >> 5;
    const int lane  = tid & 31;
    const int group = lane >> 4;
    const int ln    = lane & 15;

    const int bidx  = blockIdx.x;
    const int qtile = bidx & 63;
    const int h     = (bidx >> 6) & 7;
    const int b     = bidx >> 9;

    const int   qrow  = qtile * 64 + wave * 16 + ln;
    const float scale = 0.125f;   // 1/sqrt(64), folded into Q
    const float* qptr = Q + ((size_t)(b * N_ + qrow)) * HD_ + h * D_;

    const v16h qa0 = lda_g(qptr, 0, group, scale);
    const v16h qa1 = lda_g(qptr, 32, group, scale);

    v8f o0 = {}, o1 = {}, o2 = {}, o3 = {};
    float rm[8], rl[8];
#pragma unroll
    for (int r = 0; r < 8; ++r) { rm[r] = -INFINITY; rl[r] = 0.0f; }

    // staging map: thread -> two 16B segments per tile (rows r and r+32)
    const int srow = tid >> 3;
    const int scol = (tid & 7) * 8;

    const _Float16* gKbase = Kh + ((size_t)(b * H_ + h)) * N_ * D_;
    const _Float16* gVbase = Vth + ((size_t)(b * H_ + h)) * D_ * N_;

    auto issue_stage = [&](int nc, int buf) {
        const _Float16* gK = gKbase + (size_t)(nc * 64) * D_;
        const _Float16* gV = gVbase + nc * 64;
        _Float16* KB = &Kl[buf][0];
        _Float16* VB = &Vl[buf][0];
        stage16(KB + srow * LPAD + scol,        gK + (size_t)srow * D_ + scol);
        stage16(KB + (srow + 32) * LPAD + scol, gK + (size_t)(srow + 32) * D_ + scol);
        stage16(VB + srow * LPAD + scol,        gV + (size_t)srow * N_ + scol);
        stage16(VB + (srow + 32) * LPAD + scol, gV + (size_t)(srow + 32) * N_ + scol);
    };

    issue_stage(0, 0);   // prologue

    for (int nc = 0; nc < N_ / 64; ++nc) {
        const int cur = nc & 1;
        // chunk nc's copies were issued last iteration (or prologue)
        stage_wait();          // own async ops complete
        __syncthreads();       // everyone's complete; buf[cur^1] reads all retired
        if (nc + 1 < N_ / 64)
            issue_stage(nc + 1, cur ^ 1);   // overlaps with compute below
        if (lane == 0 && nc + 2 < N_ / 64)
            __builtin_prefetch(gKbase + (size_t)((nc + 2) * 64) * D_, 0, 0);

        const _Float16* Kbuf = &Kl[cur][0];
        const _Float16* Vbuf = &Vl[cur][0];

        // ---- S = (Q*scale) . K^T : four 16x16 tiles over 64 keys ----
        v8f st[4];
#pragma unroll
        for (int t = 0; t < 4; ++t) {
            const _Float16* kb = Kbuf + (t * 16 + ln) * LPAD;
            v16h b0 = ldb_l(kb + group * 16);
            v16h b1 = ldb_l(kb + 32 + group * 16);
            v8f acc = {};
            acc = wmma16(qa0, b0, acc);
            acc = wmma16(qa1, b1, acc);
            st[t] = acc;
        }

        // ---- online softmax: row r lives on VGPR r across a 16-lane group ----
#pragma unroll
        for (int r = 0; r < 8; ++r) {
            float mx = fmaxf(fmaxf(st[0][r], st[1][r]), fmaxf(st[2][r], st[3][r]));
            mx = red16_max(mx);
            const float mnew = fmaxf(rm[r], mx);
            const float corr = __expf(rm[r] - mnew);
            float rs = 0.0f;
#pragma unroll
            for (int t = 0; t < 4; ++t) {
                float p = __expf(st[t][r] - mnew);
                st[t][r] = p;
                rs += p;
            }
            rs = red16_sum(rs);
            rl[r] = rl[r] * corr + rs;
            rm[r] = mnew;
            o0[r] *= corr; o1[r] *= corr; o2[r] *= corr; o3[r] *= corr;
        }

        // ---- transpose P from C-layout to A-layout via per-wave LDS ----
        _Float16* prow = Pl + wave * 16 * LPAD;
#pragma unroll
        for (int t = 0; t < 4; ++t)
#pragma unroll
            for (int r = 0; r < 8; ++r)
                prow[(r + group * 8) * LPAD + t * 16 + ln] = (_Float16)st[t][r];
        // same-wave DS ops are in-order: reload immediately, no barrier needed
        const _Float16* pr = prow + ln * LPAD;
        const v16h pa0 = lda_l(pr, 0, group);
        const v16h pa1 = lda_l(pr, 32, group);

        // ---- O += P . V ----
#pragma unroll
        for (int d = 0; d < 4; ++d) {
            const _Float16* vb = Vbuf + (d * 16 + ln) * LPAD;
            v16h b0 = ldb_l(vb + group * 16);
            v16h b1 = ldb_l(vb + 32 + group * 16);
            v8f acc = (d == 0) ? o0 : (d == 1) ? o1 : (d == 2) ? o2 : o3;
            acc = wmma16(pa0, b0, acc);
            acc = wmma16(pa1, b1, acc);
            if (d == 0) o0 = acc; else if (d == 1) o1 = acc;
            else if (d == 2) o2 = acc; else o3 = acc;
        }
    }

    // ---- normalize and write context [B,N,H*D] f32 ----
    float* cp = ctx + ((size_t)(b * N_ + qtile * 64 + wave * 16)) * HD_ + h * D_;
#pragma unroll
    for (int r = 0; r < 8; ++r) {
        const int m    = r + group * 8;
        const float inv = 1.0f / rl[r];
        cp[(size_t)m * HD_ + 0 * 16 + ln] = o0[r] * inv;
        cp[(size_t)m * HD_ + 1 * 16 + ln] = o1[r] * inv;
        cp[(size_t)m * HD_ + 2 * 16 + ln] = o2[r] * inv;
        cp[(size_t)m * HD_ + 3 * 16 + ln] = o3[r] * inv;
    }
}

// ---------------------------------------------------------------------------
// Kernel 2: out[8192,64] = ctx[8192,512] @ W[512,64] + b  (WMMA f16)
// ---------------------------------------------------------------------------
__global__ __launch_bounds__(128) void mha_proj_kernel(
    const float* __restrict__ ctx, const float* __restrict__ W,
    const float* __restrict__ bias, float* __restrict__ out) {
    __shared__ _Float16 Wt[64 * LPAD];   // Wt[col][k_local]

    const int tid   = threadIdx.x;
    const int wave  = tid >> 5;
    const int lane  = tid & 31;
    const int group = lane >> 4;
    const int ln    = lane & 15;

    const size_t rowbase = (size_t)blockIdx.x * 64;
    const float* arow = ctx + (rowbase + wave * 16 + ln) * HD_;

    v8f c0 = {}, c1 = {}, c2 = {}, c3 = {};

    for (int kc = 0; kc < HD_ / 64; ++kc) {
        __syncthreads();
        {   // cooperative transpose of a 64x64 W chunk into LDS (f16)
            const int n    = (tid & 15) * 4;
            const int krow = tid >> 4;
#pragma unroll
            for (int i = 0; i < 8; ++i) {
                const int kl = krow + 8 * i;
                const int k  = kc * 64 + kl;
                float4 wf = *(const float4*)(W + (size_t)k * D_ + n);
                Wt[(n + 0) * LPAD + kl] = (_Float16)wf.x;
                Wt[(n + 1) * LPAD + kl] = (_Float16)wf.y;
                Wt[(n + 2) * LPAD + kl] = (_Float16)wf.z;
                Wt[(n + 3) * LPAD + kl] = (_Float16)wf.w;
            }
        }
        __syncthreads();

        const v16h a0 = lda_g(arow, kc * 64 + 0, group, 1.0f);
        const v16h a1 = lda_g(arow, kc * 64 + 32, group, 1.0f);
#pragma unroll
        for (int d = 0; d < 4; ++d) {
            const _Float16* wb = Wt + (d * 16 + ln) * LPAD;
            v16h b0 = ldb_l(wb + group * 16);
            v16h b1 = ldb_l(wb + 32 + group * 16);
            v8f acc = (d == 0) ? c0 : (d == 1) ? c1 : (d == 2) ? c2 : c3;
            acc = wmma16(a0, b0, acc);
            acc = wmma16(a1, b1, acc);
            if (d == 0) c0 = acc; else if (d == 1) c1 = acc;
            else if (d == 2) c2 = acc; else c3 = acc;
        }
    }

    float* orow = out + (rowbase + wave * 16) * D_;
#pragma unroll
    for (int d = 0; d < 4; ++d) {
        const float bv = bias[d * 16 + ln];
        const v8f acc = (d == 0) ? c0 : (d == 1) ? c1 : (d == 2) ? c2 : c3;
#pragma unroll
        for (int r = 0; r < 8; ++r)
            orow[(size_t)(r + group * 8) * D_ + d * 16 + ln] = acc[r] + bv;
    }
}

extern "C" void kernel_launch(void* const* d_in, const int* in_sizes, int n_in,
                              void* d_out, int out_size, void* d_ws, size_t ws_size,
                              hipStream_t stream) {
    (void)in_sizes; (void)n_in; (void)out_size; (void)ws_size;
    const float* q  = (const float*)d_in[0];
    const float* k  = (const float*)d_in[1];
    const float* v  = (const float*)d_in[2];
    const float* W  = (const float*)d_in[3];
    const float* bo = (const float*)d_in[4];

    // workspace layout: ctx f32 (16.8MB) | Kh f16 (8.4MB) | Vt f16 (8.4MB)
    char* ws = (char*)d_ws;
    float*    ctx = (float*)ws;
    _Float16* Kh  = (_Float16*)(ws + (size_t)B_ * N_ * HD_ * sizeof(float));
    _Float16* Vt  = Kh + (size_t)B_ * H_ * N_ * D_;
    float*    out = (float*)d_out;

    // 0) one-time f32 -> f16 (+ V transpose) layout prep
    mha_cvt_kernel<<<dim3((B_ * N_ * HD_ / 4) / 256), 256, 0, stream>>>(k, v, Kh, Vt);
    // 1) flash attention: B*H*(N/64) = 1024 blocks
    mha_flash_attn_kernel<<<dim3(B_ * H_ * (N_ / 64)), 256, 0, stream>>>(q, Kh, Vt, ctx);
    // 2) output projection: 128 blocks
    mha_proj_kernel<<<dim3((B_ * N_) / 64), 128, 0, stream>>>(ctx, W, bo, out);
}
